// REFT_Psi_3994319585604
// MI455X (gfx1250) — matrix operand order repaired and checked by reading, not compiled, not run on hardware
//
#include <hip/hip_runtime.h>

#define BATCH   65536
#define DIM     64
#define HID     128
#define CTXD    384
#define NPRIMES 5
#define EPSF    1e-8f

typedef __attribute__((ext_vector_type(16))) __bf16 v16bf;
typedef __attribute__((ext_vector_type(8)))  __bf16 v8bf;
typedef __attribute__((ext_vector_type(8)))  float  v8f;
typedef __attribute__((ext_vector_type(4)))  float  v4f;

__device__ __forceinline__ v8f wmma_bf16(v16bf a, v16bf b, v8f c) {
  return __builtin_amdgcn_wmma_f32_16x16x32_bf16(false, a, false, b, (short)0, c,
                                                 false, false);
}

// A-fragment (16x32 bf16): contiguous k order matching the packed B:
// lane's 16 elems are rowptr[16*hi .. +15].
__device__ __forceinline__ v16bf a_frag_f32(const float* __restrict__ rowptr, int hi) {
  const v4f* r0 = (const v4f*)(rowptr + 16 * hi);       // 32B aligned
  const v4f* r1 = (const v4f*)(rowptr + 16 * hi + 8);
  v4f x0 = r0[0], x1 = r0[1], x2 = r1[0], x3 = r1[1];
  v16bf a;
#pragma unroll
  for (int e = 0; e < 4; ++e) {
    a[e]      = (__bf16)x0[e];
    a[e + 4]  = (__bf16)x1[e];
    a[e + 8]  = (__bf16)x2[e];
    a[e + 12] = (__bf16)x3[e];
  }
  return a;
}
__device__ __forceinline__ v16bf a_frag_bf16(const __bf16* __restrict__ rowptr, int hi) {
  const v8bf* p0 = (const v8bf*)(rowptr + 16 * hi);     // 16B aligned
  const v8bf* p1 = (const v8bf*)(rowptr + 16 * hi + 8);
  v8bf x0 = p0[0], x1 = p1[0];
  v16bf a;
#pragma unroll
  for (int e = 0; e < 8; ++e) { a[e] = x0[e]; a[e + 8] = x1[e]; }
  return a;
}
// B-fragment from fragment-major packed weights: lane's 16 bf16 contiguous.
__device__ __forceinline__ v16bf b_frag_packed(const __bf16* __restrict__ packed,
                                               int tile, int lane) {
  return *(const v16bf*)(packed + ((size_t)tile * 32 + lane) * 16);  // 32B aligned
}

__device__ __forceinline__ float sigmoidf(float x) { return 1.f / (1.f + __expf(-x)); }

// ---------------------------------------------------------------------------
// Kernel 0: batch-independent scalar path + fragment-major bf16 weight prepack
// Packed layout: idx = ((tile*32 + lane)*16 + e), tile = kt*NT + t,
//   k = kt*32 + 16*(lane>>4) + e, n = t*16 + (lane&15), value = W[n*Ksrc + k]
// ---------------------------------------------------------------------------
__global__ void prep_kernel(
    const float* __restrict__ time_weights, const float* __restrict__ memory,
    const float* __restrict__ in_proj_w, const float* __restrict__ in_proj_b,
    const float* __restrict__ out_w, const float* __restrict__ out_b,
    const float* __restrict__ ctx_w, const float* __restrict__ emo_w1,
    const float* __restrict__ emo_w2,
    const float* __restrict__ wih_f, const float* __restrict__ whh_f,
    const float* __restrict__ wih_r, const float* __restrict__ whh_r,
    float* __restrict__ o_vec,
    __bf16* __restrict__ ctx_wB, __bf16* __restrict__ emo_w1B,
    __bf16* __restrict__ emo_w2B,
    __bf16* __restrict__ wihB_f, __bf16* __restrict__ whhB_f,
    __bf16* __restrict__ wihB_r, __bf16* __restrict__ whhB_r)
{
  const int gid = blockIdx.x * blockDim.x + threadIdx.x;
  const int gsz = gridDim.x * blockDim.x;

  // ctx_w [64][384] -> 12 kt x 4 t tiles
  for (int i = gid; i < 48 * 512; i += gsz) {
    const int e = i & 15, lane = (i >> 4) & 31, tile = i >> 9;
    const int kt = tile >> 2, t = tile & 3;
    const int k = (kt << 5) + ((lane >> 4) << 4) + e;
    const int n = (t << 4) + (lane & 15);
    ctx_wB[i] = (__bf16)ctx_w[n * CTXD + k];
  }
  // emo_w1 [128][64] -> 2 kt x 8 t
  for (int i = gid; i < 16 * 512; i += gsz) {
    const int e = i & 15, lane = (i >> 4) & 31, tile = i >> 9;
    const int kt = tile >> 3, t = tile & 7;
    const int k = (kt << 5) + ((lane >> 4) << 4) + e;
    const int n = (t << 4) + (lane & 15);
    emo_w1B[i] = (__bf16)emo_w1[n * DIM + k];
  }
  // emo_w2 [64][128] -> 4 kt x 4 t
  for (int i = gid; i < 16 * 512; i += gsz) {
    const int e = i & 15, lane = (i >> 4) & 31, tile = i >> 9;
    const int kt = tile >> 2, t = tile & 3;
    const int k = (kt << 5) + ((lane >> 4) << 4) + e;
    const int n = (t << 4) + (lane & 15);
    emo_w2B[i] = (__bf16)emo_w2[n * HID + k];
  }
  // gru w_ih [192][130] (K padded to 160) -> 5 kt x 12 t
  for (int i = gid; i < 60 * 512; i += gsz) {
    const int e = i & 15, lane = (i >> 4) & 31, tile = i >> 9;
    const int kt = tile / 12, t = tile % 12;
    const int k = (kt << 5) + ((lane >> 4) << 4) + e;
    const int n = (t << 4) + (lane & 15);
    const float vf = (k < 130) ? wih_f[n * 130 + k] : 0.f;
    const float vr = (k < 130) ? wih_r[n * 130 + k] : 0.f;
    wihB_f[i] = (__bf16)vf;
    wihB_r[i] = (__bf16)vr;
  }
  // gru w_hh [192][64] -> 2 kt x 12 t
  for (int i = gid; i < 24 * 512; i += gsz) {
    const int e = i & 15, lane = (i >> 4) & 31, tile = i >> 9;
    const int kt = tile / 12, t = tile % 12;
    const int k = (kt << 5) + ((lane >> 4) << 4) + e;
    const int n = (t << 4) + (lane & 15);
    whhB_f[i] = (__bf16)whh_f[n * DIM + k];
    whhB_r[i] = (__bf16)whh_r[n * DIM + k];
  }

  if (blockIdx.x == 0) {
    __shared__ float s_v[DIM];
    const int t = threadIdx.x;
    if (t < DIM) {
      float mx = time_weights[0];
      for (int m = 1; m < 10; ++m) mx = fmaxf(mx, time_weights[m]);
      float ex[10], se = 0.f;
      for (int m = 0; m < 10; ++m) { ex[m] = expf(time_weights[m] - mx); se += ex[m]; }
      const float inv = 1.f / se;
      float acc = in_proj_b[2 * DIM + t];
      for (int k = 0; k < DIM; ++k) {
        float mt = 0.f;
        for (int m = 0; m < 10; ++m) mt += ex[m] * memory[m * DIM + k];
        acc += in_proj_w[(2 * DIM + t) * DIM + k] * (mt * inv);
      }
      s_v[t] = acc;
    }
    __syncthreads();
    if (t < DIM) {
      float o = out_b[t];
      for (int d = 0; d < DIM; ++d) o += out_w[t * DIM + d] * s_v[d];
      o_vec[t] = o;
    }
  }
}

// ---------------------------------------------------------------------------
// Kernel 1: ctx GEMM + prime softmax/ideal + res/omega/stress + emotion MLP
// 4 waves/block, 16 rows/wave -> 64 rows/block, 1024 blocks
// ---------------------------------------------------------------------------
__global__ __launch_bounds__(128) void phase1_kernel(
    const float* __restrict__ theta, const float* __restrict__ context,
    const float* __restrict__ ctx_b,
    const float* __restrict__ ideal_w, const float* __restrict__ ideal_b,
    const float* __restrict__ prime_embeds, const float* __restrict__ prev_ideal,
    const float* __restrict__ freq_w, const float* __restrict__ freq_b,
    const float* __restrict__ emo_b1, const float* __restrict__ emo_b2,
    const __bf16* __restrict__ ctx_wB, const __bf16* __restrict__ emo_w1B,
    const __bf16* __restrict__ emo_w2B,
    float* __restrict__ out_e, float* __restrict__ out_res,
    float* __restrict__ out_omega, float* __restrict__ out_pw,
    float* __restrict__ partials)
{
  __shared__ float  s_ctx[4][16][DIM];
  __shared__ __bf16 s_h1[4][16][HID];
  __shared__ float  s_red[128];

  const int tid  = threadIdx.x;
  const int lane = tid & 31;
  const int wv   = tid >> 5;
  const int col  = lane & 15;
  const int hi   = lane >> 4;
  const int row0 = (blockIdx.x * 4 + wv) * 16;

  // ---- ctx = context @ ctx_w^T + ctx_b  (K=384, N=64) ----
  {
    const v8f vz = {};
    v8f acc[4] = {vz, vz, vz, vz};
    const float* arow = context + (size_t)(row0 + col) * CTXD;
#pragma unroll 4
    for (int kt = 0; kt < 12; ++kt) {
      if (kt + 2 < 12) __builtin_prefetch(arow + (kt + 2) * 32, 0, 0);
      v16bf a = a_frag_f32(arow + kt * 32, hi);
#pragma unroll
      for (int t = 0; t < 4; ++t)
        acc[t] = wmma_bf16(a, b_frag_packed(ctx_wB, kt * 4 + t, lane), acc[t]);
    }
#pragma unroll
    for (int t = 0; t < 4; ++t) {
      const float bb = ctx_b[t * 16 + col];
#pragma unroll
      for (int j = 0; j < 8; ++j)
        s_ctx[wv][j + 8 * hi][t * 16 + col] = acc[t][j] + bb;
    }
  }
  __syncthreads();

  // ---- per-row scalar path: prime softmax, theta_ideal, res/omega/stress ----
  float my_stress = 0.f;
  if (lane < 16) {
    const int r = lane;
    const int grow = row0 + r;
    const float* th = theta + (size_t)grow * DIM;
    float logit[NPRIMES];
#pragma unroll
    for (int p = 0; p < NPRIMES; ++p) {
      float acc = ideal_b[p];
      const float* wp = ideal_w + p * 2 * DIM;
      for (int d = 0; d < DIM; ++d) acc += th[d] * wp[d];
      for (int d = 0; d < DIM; ++d) acc += s_ctx[wv][r][d] * wp[DIM + d];
      logit[p] = acc;
    }
    float mx = logit[0];
#pragma unroll
    for (int p = 1; p < NPRIMES; ++p) mx = fmaxf(mx, logit[p]);
    float se = 0.f, pw[NPRIMES];
#pragma unroll
    for (int p = 0; p < NPRIMES; ++p) { pw[p] = expf(logit[p] - mx); se += pw[p]; }
    const float inv = 1.f / se;
#pragma unroll
    for (int p = 0; p < NPRIMES; ++p) {
      pw[p] *= inv;
      out_pw[(size_t)grow * NPRIMES + p] = pw[p];
    }
    float nt2 = 0.f;
    for (int d = 0; d < DIM; ++d) nt2 += th[d] * th[d];
    float ni2 = 0.f, fr2 = 0.f, dp = 0.f, st = 0.f;
    for (int d = 0; d < DIM; ++d) {
      float ti = 0.f;
#pragma unroll
      for (int p = 0; p < NPRIMES; ++p) ti += pw[p] * prime_embeds[p * DIM + d];
      ni2 += ti * ti;
      const float df = th[d] - ti;         st  += df * df;
      const float fd = ti - prev_ideal[d]; fr2 += fd * fd;
      dp += th[d] * ti;
    }
    my_stress = st;
    const float nt = sqrtf(nt2), ni = sqrtf(ni2);
    const float it = 1.f / (nt + EPSF), ii = 1.f / (ni + EPSF);
    out_res[grow] = (dp * it * ii) / fmaxf((nt * it) * (ni * ii), EPSF);
    const float freq = sqrtf(fr2);
    out_omega[grow] = tanhf(freq * freq_w[0] + freq_b[0]) * (1.f + 0.1f * sinf(freq));
  }
  s_red[tid] = my_stress;
  __syncthreads();
  if (tid == 0) {
    float s = 0.f;
    for (int i = 0; i < 128; ++i) s += s_red[i];   // fixed order -> deterministic
    partials[blockIdx.x] = s;
  }

  // ---- emotion MLP stage 1: h1 = tanh(theta @ emo_w1^T + b1), K=64 N=128 ----
  {
    const v8f vz = {};
    v8f acc[8] = {vz, vz, vz, vz, vz, vz, vz, vz};
    const float* arow = theta + (size_t)(row0 + col) * DIM;
#pragma unroll
    for (int kt = 0; kt < 2; ++kt) {
      v16bf a = a_frag_f32(arow + kt * 32, hi);
#pragma unroll
      for (int t = 0; t < 8; ++t)
        acc[t] = wmma_bf16(a, b_frag_packed(emo_w1B, kt * 8 + t, lane), acc[t]);
    }
#pragma unroll
    for (int t = 0; t < 8; ++t) {
      const float bb = emo_b1[t * 16 + col];
#pragma unroll
      for (int j = 0; j < 8; ++j)
        s_h1[wv][j + 8 * hi][t * 16 + col] = (__bf16)tanhf(acc[t][j] + bb);
    }
  }
  __syncthreads();

  // ---- emotion MLP stage 2: e = relu(h1 @ emo_w2^T + b2), K=128 N=64 ----
  {
    const v8f vz = {};
    v8f acc[4] = {vz, vz, vz, vz};
#pragma unroll
    for (int kt = 0; kt < 4; ++kt) {
      v16bf a = a_frag_bf16(&s_h1[wv][col][kt * 32], hi);
#pragma unroll
      for (int t = 0; t < 4; ++t)
        acc[t] = wmma_bf16(a, b_frag_packed(emo_w2B, kt * 4 + t, lane), acc[t]);
    }
#pragma unroll
    for (int t = 0; t < 4; ++t) {
      const float bb = emo_b2[t * 16 + col];
#pragma unroll
      for (int j = 0; j < 8; ++j)
        out_e[(size_t)(row0 + j + 8 * hi) * DIM + t * 16 + col] =
            fmaxf(0.f, acc[t][j] + bb);   // raw e; stress added in phase2
    }
  }
}

// ---------------------------------------------------------------------------
// Kernel 2: deterministic global stress reduction
// ---------------------------------------------------------------------------
__global__ void stress_reduce_kernel(const float* __restrict__ partials,
                                     float* __restrict__ total) {
  __shared__ float s[256];
  const int t = threadIdx.x;
  float v = 0.f;
  for (int i = t; i < 1024; i += 256) v += partials[i];
  s[t] = v;
  __syncthreads();
  for (int off = 128; off > 0; off >>= 1) {
    if (t < off) s[t] += s[t + off];
    __syncthreads();
  }
  if (t == 0) total[0] = s[0];
}

// ---------------------------------------------------------------------------
// Kernel 3: bidirectional GRU (K padded to 160), theta_next/h_next/e_theta
// ---------------------------------------------------------------------------
__global__ __launch_bounds__(128) void phase2_kernel(
    const float* __restrict__ theta, const float* __restrict__ h_prev,
    const float* __restrict__ o_vec, const float* __restrict__ stress_total,
    const __bf16* __restrict__ wihB_f, const __bf16* __restrict__ whhB_f,
    const __bf16* __restrict__ wihB_r, const __bf16* __restrict__ whhB_r,
    const float* __restrict__ bih_f, const float* __restrict__ bhh_f,
    const float* __restrict__ bih_r, const float* __restrict__ bhh_r,
    const float* __restrict__ res_in, const float* __restrict__ omega_in,
    float* __restrict__ out_e,
    float* __restrict__ out_theta_next, float* __restrict__ out_hnext)
{
  __shared__ __bf16 s_xt[4][16][160];
  const int tid  = threadIdx.x;
  const int lane = tid & 31;
  const int wv   = tid >> 5;
  const int col  = lane & 15;
  const int hi   = lane >> 4;
  const int row0 = (blockIdx.x * 4 + wv) * 16;
  const float sadd = 0.001f * stress_total[0];

  // build x_t = [o_vec | e+0.001*stress | res | omega | zero-pad] as bf16
  {
    const int r = col;
    const int grow = row0 + r;
    if (hi == 0) {
      for (int c = 0; c < DIM; ++c) s_xt[wv][r][c] = (__bf16)o_vec[c];
      for (int d = 0; d < 16; ++d) {
        const float ef = out_e[(size_t)grow * DIM + d] + sadd;
        out_e[(size_t)grow * DIM + d] = ef;
        s_xt[wv][r][DIM + d] = (__bf16)ef;
      }
    } else {
      for (int d = 16; d < DIM; ++d) {
        const float ef = out_e[(size_t)grow * DIM + d] + sadd;
        out_e[(size_t)grow * DIM + d] = ef;
        s_xt[wv][r][DIM + d] = (__bf16)ef;
      }
      s_xt[wv][r][128] = (__bf16)res_in[grow];
      s_xt[wv][r][129] = (__bf16)omega_in[grow];
      for (int c = 130; c < 160; ++c) s_xt[wv][r][c] = (__bf16)0.f;
    }
  }
  __syncthreads();

  for (int dir = 0; dir < 2; ++dir) {
    const __bf16* wih = dir ? wihB_r : wihB_f;
    const __bf16* whh = dir ? whhB_r : whhB_f;
    const float*  bih = dir ? bih_r  : bih_f;
    const float*  bhh = dir ? bhh_r  : bhh_f;
    const float*  hbase = h_prev + (size_t)dir * BATCH * DIM;

    const v8f vz = {};
    v8f acc[12];
#pragma unroll
    for (int t = 0; t < 12; ++t) acc[t] = vz;

    // gi = x_t @ w_ih^T   (K=160, N=192)
#pragma unroll
    for (int kt = 0; kt < 5; ++kt) {
      v16bf a = a_frag_bf16(&s_xt[wv][col][kt * 32], hi);
#pragma unroll
      for (int t = 0; t < 12; ++t)
        acc[t] = wmma_bf16(a, b_frag_packed(wih, kt * 12 + t, lane), acc[t]);
    }
    // gh = h @ w_hh^T: r,z gates accumulate onto gi; n-gate kept separate
    v8f acch[4] = {vz, vz, vz, vz};
    const float* hrow = hbase + (size_t)(row0 + col) * DIM;
#pragma unroll
    for (int kt = 0; kt < 2; ++kt) {
      v16bf a = a_frag_f32(hrow + kt * 32, hi);
#pragma unroll
      for (int t = 0; t < 8; ++t)
        acc[t] = wmma_bf16(a, b_frag_packed(whh, kt * 12 + t, lane), acc[t]);
#pragma unroll
      for (int t = 0; t < 4; ++t)
        acch[t] = wmma_bf16(a, b_frag_packed(whh, kt * 12 + 8 + t, lane), acch[t]);
    }
    // gates — lane/VGPR mapping identical across N-tiles, all in registers
#pragma unroll
    for (int t = 0; t < 4; ++t) {
      const int d = t * 16 + col;
      const float br   = bih[d] + bhh[d];
      const float bz   = bih[DIM + d] + bhh[DIM + d];
      const float bn_i = bih[2 * DIM + d];
      const float bn_h = bhh[2 * DIM + d];
#pragma unroll
      for (int j = 0; j < 8; ++j) {
        const int grow = row0 + j + 8 * hi;
        const float rg = sigmoidf(acc[t][j] + br);
        const float zg = sigmoidf(acc[t + 4][j] + bz);
        const float ng = tanhf(acc[t + 8][j] + bn_i + rg * (acch[t][j] + bn_h));
        const float hp = hbase[(size_t)grow * DIM + d];
        const float hn = (1.f - zg) * ng + zg * hp;
        out_hnext[(size_t)dir * BATCH * DIM + (size_t)grow * DIM + d] = hn;
        if (dir == 0)
          out_theta_next[(size_t)grow * DIM + d] =
              0.3f * theta[(size_t)grow * DIM + d] + 0.7f * hn;
      }
    }
  }
}

// ---------------------------------------------------------------------------
extern "C" void kernel_launch(void* const* d_in, const int* in_sizes, int n_in,
                              void* d_out, int out_size, void* d_ws, size_t ws_size,
                              hipStream_t stream) {
  (void)in_sizes; (void)n_in; (void)out_size; (void)ws_size;
  const float* theta        = (const float*)d_in[0];
  const float* context      = (const float*)d_in[1];
  const float* h_prev       = (const float*)d_in[2];
  const float* memory       = (const float*)d_in[3];
  const float* time_weights = (const float*)d_in[4];
  const float* in_proj_w    = (const float*)d_in[5];
  const float* in_proj_b    = (const float*)d_in[6];
  const float* out_w        = (const float*)d_in[7];
  const float* out_b        = (const float*)d_in[8];
  const float* emo_w1       = (const float*)d_in[9];
  const float* emo_b1       = (const float*)d_in[10];
  const float* emo_w2       = (const float*)d_in[11];
  const float* emo_b2       = (const float*)d_in[12];
  const float* prime_embeds = (const float*)d_in[13];
  const float* ctx_w        = (const float*)d_in[14];
  const float* ctx_b        = (const float*)d_in[15];
  const float* ideal_w      = (const float*)d_in[16];
  const float* ideal_b      = (const float*)d_in[17];
  const float* gru_w_ih_f   = (const float*)d_in[18];
  const float* gru_w_hh_f   = (const float*)d_in[19];
  const float* gru_b_ih_f   = (const float*)d_in[20];
  const float* gru_b_hh_f   = (const float*)d_in[21];
  const float* gru_w_ih_r   = (const float*)d_in[22];
  const float* gru_w_hh_r   = (const float*)d_in[23];
  const float* gru_b_ih_r   = (const float*)d_in[24];
  const float* gru_b_hh_r   = (const float*)d_in[25];
  const float* freq_w       = (const float*)d_in[26];
  const float* freq_b       = (const float*)d_in[27];
  const float* prev_ideal   = (const float*)d_in[28];

  // Output layout: theta_next | h_next(2) | e_theta | res | omega | prime_weights
  float* out            = (float*)d_out;
  float* out_theta_next = out;
  float* out_hnext      = out + (size_t)BATCH * DIM;
  float* out_e          = out + (size_t)3 * BATCH * DIM;
  float* out_res        = out + (size_t)4 * BATCH * DIM;
  float* out_omega      = out_res + BATCH;
  float* out_pw         = out_omega + BATCH;

  // Workspace carve-up (floats padded to 1096 so bf16 arrays are 32B-aligned;
  // each packed array is a multiple of 512 elems = 1KB, preserving alignment)
  float* ws_f         = (float*)d_ws;
  float* o_vec        = ws_f;                 // 64
  float* partials     = ws_f + 64;            // 1024
  float* stress_total = ws_f + 64 + 1024;     // 1 (+7 pad)
  __bf16* wb      = (__bf16*)(ws_f + 1096);
  __bf16* ctx_wB  = wb;  wb += 48 * 512;      // 12 kt x 4 t
  __bf16* emo_w1B = wb;  wb += 16 * 512;      // 2 kt x 8 t
  __bf16* emo_w2B = wb;  wb += 16 * 512;      // 4 kt x 4 t
  __bf16* wihB_f  = wb;  wb += 60 * 512;      // 5 kt x 12 t (K padded 130->160)
  __bf16* whhB_f  = wb;  wb += 24 * 512;      // 2 kt x 12 t
  __bf16* wihB_r  = wb;  wb += 60 * 512;
  __bf16* whhB_r  = wb;  wb += 24 * 512;

  prep_kernel<<<128, 256, 0, stream>>>(
      time_weights, memory, in_proj_w, in_proj_b, out_w, out_b,
      ctx_w, emo_w1, emo_w2,
      gru_w_ih_f, gru_w_hh_f, gru_w_ih_r, gru_w_hh_r,
      o_vec, ctx_wB, emo_w1B, emo_w2B, wihB_f, whhB_f, wihB_r, whhB_r);

  phase1_kernel<<<BATCH / 64, 128, 0, stream>>>(
      theta, context, ctx_b, ideal_w, ideal_b, prime_embeds, prev_ideal,
      freq_w, freq_b, emo_b1, emo_b2, ctx_wB, emo_w1B, emo_w2B,
      out_e, out_res, out_omega, out_pw, partials);

  stress_reduce_kernel<<<1, 256, 0, stream>>>(partials, stress_total);

  phase2_kernel<<<BATCH / 64, 128, 0, stream>>>(
      theta, h_prev, o_vec, stress_total,
      wihB_f, whhB_f, wihB_r, whhB_r,
      gru_b_ih_f, gru_b_hh_f, gru_b_ih_r, gru_b_hh_r,
      out_res, out_omega, out_e, out_theta_next, out_hnext);
}
